// DeepSQN_71829033058724
// MI455X (gfx1250) — matrix operand order repaired and checked by reading, not compiled, not run on hardware
//
#include <hip/hip_runtime.h>
#include <hip/hip_bf16.h>

// ---------------------------------------------------------------------------
// Types for CDNA5 WMMA (gfx1250, wave32)
// ---------------------------------------------------------------------------
typedef __attribute__((ext_vector_type(16))) __bf16          v16bf;
typedef __attribute__((ext_vector_type(8)))  float           v8f;
typedef __attribute__((ext_vector_type(8)))  unsigned short  su8;
typedef __attribute__((ext_vector_type(4)))  int             v4i;

typedef __attribute__((address_space(1))) v4i* gv4i_p;   // global-memory v4i*
typedef __attribute__((address_space(3))) v4i* lv4i_p;   // LDS v4i*

union AFrag {
    v16bf          v;
    su8            h[2];
    unsigned int   d[8];
    unsigned short u[16];
};

__device__ __forceinline__ unsigned short f2bf(float f) {
    unsigned int u = __float_as_uint(f);
    unsigned int lsb = (u >> 16) & 1u;
    u += 0x7FFFu + lsb;                 // round to nearest even
    return (unsigned short)(u >> 16);
}

#if __has_builtin(__builtin_amdgcn_global_load_async_to_lds_b128)
#define HAVE_ASYNC_COPY 1
#else
#define HAVE_ASYNC_COPY 0
#endif

__device__ __forceinline__ void wait_async0() {
#if __has_builtin(__builtin_amdgcn_s_wait_asynccnt)
    __builtin_amdgcn_s_wait_asynccnt(0);
#else
    asm volatile("s_wait_asynccnt 0x0" ::: "memory");
#endif
}

// ---------------------------------------------------------------------------
// Problem constants
// ---------------------------------------------------------------------------
#define T      8
#define BATCH  128
// layer1: conv 4->32, 8x8, s4 : 84x84 -> 20x20
#define C1 32
#define HW1 400            // 20*20
#define NE1 (BATCH*C1*HW1) // 1,638,400
// layer2: conv 32->64, 4x4, s2 : 20x20 -> 9x9  (K = 32*16 = 512)
#define C2 64
#define HW2 81
#define NE2 (BATCH*C2*HW2) // 663,552
#define K2  512
#define M2  (T*BATCH*HW2)  // 82,944 rows
// layer3: conv 64->64, 3x3, s1 : 9x9 -> 7x7   (K = 64*9 = 576)
#define C3 64
#define HW3 49
#define NE3 (BATCH*C3*HW3) // 401,408
#define K3  576
#define M3  (T*BATCH*HW3)  // 50,176 rows
// fc1: 3136 -> 512
#define KF  3136
#define NF  512
#define MF  (T*BATCH)      // 1024 rows

// ---------------------------------------------------------------------------
// f32 -> bf16 (weights)
// ---------------------------------------------------------------------------
__global__ void cvt_bf16_kernel(const float* __restrict__ in,
                                unsigned short* __restrict__ out, int n) {
    int i = blockIdx.x * blockDim.x + threadIdx.x;
    if (i < n) out[i] = f2bf(in[i]);
}

// ---------------------------------------------------------------------------
// conv1: direct f32 conv, computed ONCE for B=128 (input replicated over T,
// conv+BN are linear -> identical across timesteps). ~0.8 GFLOP scalar.
// ---------------------------------------------------------------------------
__global__ void conv1_kernel(const float* __restrict__ x,
                             const float* __restrict__ w,
                             float* __restrict__ h1) {
    int idx = blockIdx.x * blockDim.x + threadIdx.x;   // [128][32][20][20]
    if (idx >= NE1) return;
    int ow = idx % 20;
    int oh = (idx / 20) % 20;
    int c  = (idx / 400) % C1;
    int b  =  idx / (400 * C1);
    const float* xb = x + (size_t)b * 4 * 84 * 84;
    const float* wc = w + (size_t)c * 4 * 64;
    float acc = 0.f;
    for (int ci = 0; ci < 4; ++ci)
        for (int kh = 0; kh < 8; ++kh) {
            const float* xr = xb + (size_t)(ci * 84 + oh * 4 + kh) * 84 + ow * 4;
            const float* wr = wc + ci * 64 + kh * 8;
            #pragma unroll
            for (int kw = 0; kw < 8; ++kw) acc += xr[kw] * wr[kw];
        }
    h1[idx] = acc;
}

// ---------------------------------------------------------------------------
// Training-mode BN statistics: one block per channel.
// stats[2c] = g*rsqrt(var+eps), stats[2c+1] = b - mean*scale
// (conv bias omitted upstream: a per-channel constant cancels in batch BN)
// ---------------------------------------------------------------------------
__global__ void bn_stats_kernel(const float* __restrict__ h,
                                const float* __restrict__ g,
                                const float* __restrict__ b,
                                float* __restrict__ stats,
                                int C, int HW, int TB) {
    __shared__ float ssum[256];
    __shared__ float ssq[256];
    int c = blockIdx.x;
    int N = TB * HW;
    float sum = 0.f, sq = 0.f;
    for (int i = threadIdx.x; i < N; i += blockDim.x) {
        int tb = i / HW, hw = i - tb * HW;
        float v = h[((size_t)tb * C + c) * HW + hw];
        sum += v; sq += v * v;
    }
    ssum[threadIdx.x] = sum; ssq[threadIdx.x] = sq;
    __syncthreads();
    for (int s = 128; s > 0; s >>= 1) {
        if ((int)threadIdx.x < s) {
            ssum[threadIdx.x] += ssum[threadIdx.x + s];
            ssq[threadIdx.x]  += ssq[threadIdx.x + s];
        }
        __syncthreads();
    }
    if (threadIdx.x == 0) {
        float mean = ssum[0] / N;
        float var  = ssq[0] / N - mean * mean;
        float scale = g[c] * rsqrtf(var + 1e-5f);
        stats[2 * c]     = scale;
        stats[2 * c + 1] = b[c] - mean * scale;
    }
}

// ---------------------------------------------------------------------------
// LIF layer1: constant input per element across T -> run 8-step recurrence in
// registers, emit bf16 spike maps for all 8 timesteps.
// v' = 0.5*(v + x); s = (v' >= 1); v = v'*(1-s)
// ---------------------------------------------------------------------------
__global__ void lif1_kernel(const float* __restrict__ h1,
                            const float* __restrict__ stats,
                            unsigned short* __restrict__ s1) {
    int e = blockIdx.x * blockDim.x + threadIdx.x;
    if (e >= NE1) return;
    int c = (e / HW1) % C1;
    float xin = h1[e] * stats[2 * c] + stats[2 * c + 1];
    float v = 0.f;
    #pragma unroll
    for (int t = 0; t < T; ++t) {
        v = 0.5f * (v + xin);
        float s = (v >= 1.0f) ? 1.0f : 0.0f;
        v *= (1.0f - s);
        s1[(size_t)t * NE1 + e] = f2bf(s);
    }
}

// Generic BN+LIF for layers 2/3 (input differs per timestep).
__global__ void lif_seq_kernel(const float* __restrict__ h,
                               const float* __restrict__ stats,
                               unsigned short* __restrict__ sout,
                               int C, int HW) {
    int NE = BATCH * C * HW;
    int e = blockIdx.x * blockDim.x + threadIdx.x;
    if (e >= NE) return;
    int c = (e / HW) % C;
    float sc = stats[2 * c], sh = stats[2 * c + 1];
    float v = 0.f;
    #pragma unroll
    for (int t = 0; t < T; ++t) {
        float xin = h[(size_t)t * NE + e] * sc + sh;
        v = 0.5f * (v + xin);
        float s = (v >= 1.0f) ? 1.0f : 0.0f;
        v *= (1.0f - s);
        sout[(size_t)t * NE + e] = f2bf(s);
    }
}

// ---------------------------------------------------------------------------
// conv2 as implicit GEMM, bf16 WMMA 16x16x32, f32 accumulate.
// M = 82944 rows (tb,oh,ow), N = 64, K = 512 (k = ci*16 + kh*4 + kw).
// A-fragment structure: an 8-run of K maps to two 4-wide input rows of one
// channel, and the +16 half is the same two rows of channel ci+1
// -> 8 dword loads per WMMA instead of 16 scalar gathers.
// ---------------------------------------------------------------------------
__global__ void conv2_wmma_kernel(const unsigned short* __restrict__ s1,
                                  const unsigned short* __restrict__ w2b,
                                  float* __restrict__ h2) {
    const int MT = M2 / 16;  // 5184
    const int NT = C2 / 16;  // 4
    int wave = (blockIdx.x * blockDim.x + threadIdx.x) >> 5;
    int lane = threadIdx.x & 31;
    if (wave >= MT * NT) return;
    int tm = wave / NT, tn = wave - tm * NT;
    int col = lane & 15;
    int hi  = (lane & 16) ? 1 : 0;
    int kbA = hi ? 8 : 0;    // A: lanes 16-31 hold K 8-15 / 24-31
    int kbB = hi ? 8 : 0;    // B: lanes 16-31 hold K 8-15 / 24-31
    int n   = tn * 16 + col;

    int r0 = tm * 16 + col;              // this lane's A row (M = lane%16)
    int ow = r0 % 9;
    int oh = (r0 / 9) % 9;
    int tb = r0 / 81;
    // base pixel pointer for this output position (ih = oh*2, iw = ow*2)
    const unsigned short* inb = s1 + (size_t)tb * C1 * HW1 + (oh * 2) * 20 + ow * 2;
    const unsigned short* bp  = w2b + (size_t)n * K2;

    int kA  = kbA;                       // kk + kbA, advanced by 32 per iter
    v8f acc = {};
    for (int kk = 0; kk < K2; kk += 32, kA += 32) {
        int ci  = kA >> 4;               // channel of the low half
        int khb = (kA & 15) >> 2;        // 0 or 2 (two rows per 8-run)
        const unsigned short* p0 = inb + (size_t)ci * HW1 + khb * 20;
        const unsigned short* p1 = p0 + HW1;     // channel ci+1, same rows
        AFrag a;
        a.d[0] = *(const unsigned int*)(p0);          // kh=khb,   kw 0..1
        a.d[1] = *(const unsigned int*)(p0 + 2);      // kh=khb,   kw 2..3
        a.d[2] = *(const unsigned int*)(p0 + 20);     // kh=khb+1, kw 0..1
        a.d[3] = *(const unsigned int*)(p0 + 22);     // kh=khb+1, kw 2..3
        a.d[4] = *(const unsigned int*)(p1);          // +16 half: channel ci+1
        a.d[5] = *(const unsigned int*)(p1 + 2);
        a.d[6] = *(const unsigned int*)(p1 + 20);
        a.d[7] = *(const unsigned int*)(p1 + 22);
        AFrag bq;
        bq.h[0] = *(const su8*)(bp + kk + kbB);
        bq.h[1] = *(const su8*)(bp + kk + kbB + 16);
        v16bf bf = bq.v;
        acc = __builtin_amdgcn_wmma_f32_16x16x32_bf16(
            false, a.v, false, bf, (short)0, acc, false, false);
    }
    #pragma unroll
    for (int r = 0; r < 8; ++r) {         // C/D: VGPR r -> M = r (+8 hi half)
        int row = tm * 16 + r + hi * 8;
        int oww = row % 9, ohh = (row / 9) % 9, tbb = row / 81;
        h2[((size_t)tbb * C2 + n) * HW2 + ohh * 9 + oww] = acc[r];
    }
}

// conv3 implicit GEMM: M = 50176, N = 64, K = 576 (ci*9 + kh*3 + kw)
__global__ void conv3_wmma_kernel(const unsigned short* __restrict__ s2,
                                  const unsigned short* __restrict__ w3b,
                                  float* __restrict__ h3) {
    const int MT = M3 / 16;  // 3136
    const int NT = C3 / 16;  // 4
    int wave = (blockIdx.x * blockDim.x + threadIdx.x) >> 5;
    int lane = threadIdx.x & 31;
    if (wave >= MT * NT) return;
    int tm = wave / NT, tn = wave - tm * NT;
    int col = lane & 15;
    int hi  = (lane & 16) ? 1 : 0;
    int kbA = hi ? 16 : 0;
    int kbB = hi ? 16 : 0;
    int n   = tn * 16 + col;

    int r0 = tm * 16 + col;
    int ow = r0 % 7;
    int oh = (r0 / 7) % 7;
    int tb = r0 / 49;
    const unsigned short* inb = s2 + (size_t)tb * C2 * HW2;
    const unsigned short* bp  = w3b + (size_t)n * K3;

    v8f acc = {};
    for (int kk = 0; kk < K3; kk += 32) {
        AFrag a;
        #pragma unroll
        for (int j = 0; j < 8; ++j) {
            int k  = kk + kbA + j;
            int ci = k / 9, rem = k - ci * 9, kh = rem / 3, kw = rem - kh * 3;
            a.u[j] = inb[(size_t)(ci * 9 + oh + kh) * 9 + ow + kw];
            int k2  = k + 8;
            int ci2 = k2 / 9, rem2 = k2 - ci2 * 9, kh2 = rem2 / 3, kw2 = rem2 - kh2 * 3;
            a.u[j + 8] = inb[(size_t)(ci2 * 9 + oh + kh2) * 9 + ow + kw2];
        }
        v16bf bf = *(const v16bf*)(bp + kk + kbB);
        acc = __builtin_amdgcn_wmma_f32_16x16x32_bf16(
            false, a.v, false, bf, (short)0, acc, false, false);
    }
    #pragma unroll
    for (int r = 0; r < 8; ++r) {
        int row = tm * 16 + r + hi * 8;
        int oww = row % 7, ohh = (row / 7) % 7, tbb = row / 49;
        h3[((size_t)tbb * C3 + n) * HW3 + ohh * 7 + oww] = acc[r];
    }
}

// ---------------------------------------------------------------------------
// fc1 GEMM: M = 1024, N = 512, K = 3136.
// One block (8 waves) per 16-row M-tile; the 16xK A-tile is shared by all
// 32 N-tiles, so it is staged through LDS in 16x64 chunks using the CDNA5
// async global->LDS copy path (ASYNCcnt + s_wait_asynccnt), then each wave
// reads its A fragments with ds_load_b128. Each wave owns 4 N-tiles.
// ---------------------------------------------------------------------------
__global__ void __launch_bounds__(256)
fc1_wmma_kernel(const unsigned short* __restrict__ s3,
                const unsigned short* __restrict__ wfb,
                float* __restrict__ h4) {
    __shared__ unsigned short Atile[16 * 64];   // 2 KB chunk of the A tile

    int tid  = threadIdx.x;
    int wave = tid >> 5;
    int lane = tid & 31;
    int tmBase = blockIdx.x * 16;               // 64 blocks

    int col = lane & 15;
    int hi  = (lane & 16) ? 1 : 0;
    int kbA = hi ? 16 : 0;
    int kbB = hi ? 16 : 0;

    // this wave's 4 output-column tiles: n = wave*64 + i*16 + col
    const unsigned short* bp0 = wfb + (size_t)(wave * 64 + col) * KF;

    // staging assignment: threads 0..127 copy 16 rows x 64 bf16 (16B each)
    int srow = tid >> 3;                        // 0..31 (only <16 active)
    int sseg = tid & 7;                         // 8 segments of 8 bf16
    const unsigned short* gsrc = s3 + (size_t)(tmBase + srow) * KF + sseg * 8;
    unsigned short*       ldst = Atile + srow * 64 + sseg * 8;

    const unsigned short* arow = Atile + (lane & 15) * 64;

    v8f acc0 = {}, acc1 = {}, acc2 = {}, acc3 = {};
    for (int kk = 0; kk < KF; kk += 64) {       // 49 chunks
        if (tid < 128) {
#if HAVE_ASYNC_COPY
            __builtin_amdgcn_global_load_async_to_lds_b128(
                (gv4i_p)(gsrc + kk), (lv4i_p)ldst, 0, 0);
#else
            *(su8*)ldst = *(const su8*)(gsrc + kk);
#endif
        }
        // prefetch next B chunk of this wave's weight streams
        if (kk + 64 < KF) __builtin_prefetch(bp0 + kk + 64, 0, 1);
#if HAVE_ASYNC_COPY
        wait_async0();
#endif
        __syncthreads();
        #pragma unroll
        for (int sk = 0; sk < 64; sk += 32) {
            AFrag a;
            a.h[0] = *(const su8*)(arow + sk + kbA);        // K kb..kb+7
            a.h[1] = *(const su8*)(arow + sk + kbA + 8);    // K kb+8..kb+15
            v16bf b0 = *(const v16bf*)(bp0 + 0 * 16 * KF + kk + sk + kbB);
            v16bf b1 = *(const v16bf*)(bp0 + 1 * 16 * KF + kk + sk + kbB);
            v16bf b2 = *(const v16bf*)(bp0 + 2 * 16 * KF + kk + sk + kbB);
            v16bf b3 = *(const v16bf*)(bp0 + 3 * 16 * KF + kk + sk + kbB);
            acc0 = __builtin_amdgcn_wmma_f32_16x16x32_bf16(false, a.v, false, b0, (short)0, acc0, false, false);
            acc1 = __builtin_amdgcn_wmma_f32_16x16x32_bf16(false, a.v, false, b1, (short)0, acc1, false, false);
            acc2 = __builtin_amdgcn_wmma_f32_16x16x32_bf16(false, a.v, false, b2, (short)0, acc2, false, false);
            acc3 = __builtin_amdgcn_wmma_f32_16x16x32_bf16(false, a.v, false, b3, (short)0, acc3, false, false);
        }
        __syncthreads();
    }
    #pragma unroll
    for (int r = 0; r < 8; ++r) {
        int row = tmBase + r + hi * 8;
        float* op = h4 + (size_t)row * NF + wave * 64 + col;
        op[0]  = acc0[r];
        op[16] = acc1[r];
        op[32] = acc2[r];
        op[48] = acc3[r];
    }
}

// LIF on fc1 output (adds fc1 bias; no BN here)
__global__ void lif4_kernel(const float* __restrict__ h4,
                            const float* __restrict__ fc1_b,
                            float* __restrict__ s4) {
    int e = blockIdx.x * blockDim.x + threadIdx.x;   // 128*512
    if (e >= BATCH * NF) return;
    int o = e % NF;
    float bias = fc1_b[o];
    float v = 0.f;
    #pragma unroll
    for (int t = 0; t < T; ++t) {
        float xin = h4[(size_t)t * BATCH * NF + e] + bias;
        v = 0.5f * (v + xin);
        float s = (v >= 1.0f) ? 1.0f : 0.0f;
        v *= (1.0f - s);
        s4[(size_t)t * BATCH * NF + e] = s;
    }
}

// Output head: out[b][k] = mean_t sum_o s4[t][b][o] * fco_w[k][o] + fco_b[k]
__global__ void out_kernel(const float* __restrict__ s4,
                           const float* __restrict__ fco_w,
                           const float* __restrict__ fco_b,
                           float* __restrict__ out) {
    int idx = blockIdx.x * blockDim.x + threadIdx.x;  // 128*2
    if (idx >= BATCH * 2) return;
    int k = idx & 1, b = idx >> 1;
    const float* wp = fco_w + (size_t)k * NF;
    float acc = 0.f;
    for (int t = 0; t < T; ++t) {
        const float* sp = s4 + (size_t)t * BATCH * NF + (size_t)b * NF;
        for (int o = 0; o < NF; ++o) acc += sp[o] * wp[o];
    }
    out[idx] = acc * (1.0f / T) + fco_b[k];
}

// ---------------------------------------------------------------------------
// Host launcher
// ---------------------------------------------------------------------------
extern "C" void kernel_launch(void* const* d_in, const int* in_sizes, int n_in,
                              void* d_out, int out_size, void* d_ws, size_t ws_size,
                              hipStream_t stream) {
    const float* x       = (const float*)d_in[0];
    const float* conv1_w = (const float*)d_in[1];
    const float* bn1_g   = (const float*)d_in[3];
    const float* bn1_b   = (const float*)d_in[4];
    const float* conv2_w = (const float*)d_in[5];
    const float* bn2_g   = (const float*)d_in[7];
    const float* bn2_b   = (const float*)d_in[8];
    const float* conv3_w = (const float*)d_in[9];
    const float* bn3_g   = (const float*)d_in[11];
    const float* bn3_b   = (const float*)d_in[12];
    const float* fc1_w   = (const float*)d_in[13];
    const float* fc1_b   = (const float*)d_in[14];
    const float* fco_w   = (const float*)d_in[15];
    const float* fco_b   = (const float*)d_in[16];
    float* out = (float*)d_out;

    // workspace carve (256B aligned slices)
    char* ws = (char*)d_ws;
    size_t off = 0;
    auto carve = [&](size_t bytes) -> void* {
        void* p = ws + off;
        off = (off + bytes + 255) & ~(size_t)255;
        return p;
    };
    float*          h1     = (float*)carve((size_t)NE1 * 4);
    unsigned short* s1     = (unsigned short*)carve((size_t)T * NE1 * 2);
    float*          h2     = (float*)carve((size_t)T * NE2 * 4);
    unsigned short* s2     = (unsigned short*)carve((size_t)T * NE2 * 2);
    float*          h3     = (float*)carve((size_t)T * NE3 * 4);
    unsigned short* s3     = (unsigned short*)carve((size_t)T * NE3 * 2);
    float*          h4     = (float*)carve((size_t)T * BATCH * NF * 4);
    float*          s4     = (float*)carve((size_t)T * BATCH * NF * 4);
    unsigned short* w2b    = (unsigned short*)carve((size_t)C2 * K2 * 2);
    unsigned short* w3b    = (unsigned short*)carve((size_t)C3 * K3 * 2);
    unsigned short* wfb    = (unsigned short*)carve((size_t)NF * KF * 2);
    float*          stats1 = (float*)carve(2 * C1 * 4);
    float*          stats2 = (float*)carve(2 * C2 * 4);
    float*          stats3 = (float*)carve(2 * C3 * 4);
    (void)ws_size; (void)in_sizes; (void)n_in; (void)out_size;

    const int TPB = 256;

    // weight conversions f32 -> bf16
    cvt_bf16_kernel<<<(C2 * K2 + TPB - 1) / TPB, TPB, 0, stream>>>(conv2_w, w2b, C2 * K2);
    cvt_bf16_kernel<<<(C3 * K3 + TPB - 1) / TPB, TPB, 0, stream>>>(conv3_w, w3b, C3 * K3);
    cvt_bf16_kernel<<<(NF * KF + TPB - 1) / TPB, TPB, 0, stream>>>(fc1_w, wfb, NF * KF);

    // layer 1 (computed once for B=128; identical across timesteps)
    conv1_kernel<<<(NE1 + TPB - 1) / TPB, TPB, 0, stream>>>(x, conv1_w, h1);
    bn_stats_kernel<<<C1, TPB, 0, stream>>>(h1, bn1_g, bn1_b, stats1, C1, HW1, BATCH);
    lif1_kernel<<<(NE1 + TPB - 1) / TPB, TPB, 0, stream>>>(h1, stats1, s1);

    // layer 2: WMMA implicit GEMM -> BN -> LIF
    {
        int waves = (M2 / 16) * (C2 / 16);           // 20736
        conv2_wmma_kernel<<<(waves * 32 + TPB - 1) / TPB, TPB, 0, stream>>>(s1, w2b, h2);
        bn_stats_kernel<<<C2, TPB, 0, stream>>>(h2, bn2_g, bn2_b, stats2, C2, HW2, T * BATCH);
        lif_seq_kernel<<<(NE2 + TPB - 1) / TPB, TPB, 0, stream>>>(h2, stats2, s2, C2, HW2);
    }

    // layer 3: WMMA implicit GEMM -> BN -> LIF
    {
        int waves = (M3 / 16) * (C3 / 16);           // 12544
        conv3_wmma_kernel<<<(waves * 32 + TPB - 1) / TPB, TPB, 0, stream>>>(s2, w3b, h3);
        bn_stats_kernel<<<C3, TPB, 0, stream>>>(h3, bn3_g, bn3_b, stats3, C3, HW3, T * BATCH);
        lif_seq_kernel<<<(NE3 + TPB - 1) / TPB, TPB, 0, stream>>>(h3, stats3, s3, C3, HW3);
    }

    // fc1: LDS-staged WMMA GEMM (async copy path) -> LIF
    {
        fc1_wmma_kernel<<<MF / 16, 256, 0, stream>>>(s3, wfb, h4);   // 64 blocks
        lif4_kernel<<<(BATCH * NF + TPB - 1) / TPB, TPB, 0, stream>>>(h4, fc1_b, s4);
    }

    // output head + temporal mean
    out_kernel<<<1, 256, 0, stream>>>(s4, fco_w, fco_b, out);
}
